// VCSMC_63986422776301
// MI455X (gfx1250) — matrix operand order repaired and checked by reading, not compiled, not run on hardware
//
#include <hip/hip_runtime.h>
#include <hip/hip_bf16.h>
#include <math.h>

// Problem constants (match reference)
#define Kc 128
#define Jc 4
#define Nc 64
#define Rc 32
#define Tc 32
#define Sc 256
#define Ac 4
#define Dc 64
#define Zc (Kc * Jc)   // 512

// Output offsets (floats), concatenated in reference return order
constexpr long OFF_M1   = 0;                         // merge1_out  K x 33
constexpr long OFF_M2   = OFF_M1 + Kc * 33;          // merge2_out  K x 33
constexpr long OFF_B1   = OFF_M2 + Kc * 33;          // b1_all      K x 33
constexpr long OFF_B2   = OFF_B1 + Kc * 33;          // b2_all      K x 33
constexpr long OFF_EMBR = OFF_B2 + Kc * 33;          // embr_out    K x 33 x 64
constexpr long OFF_LC   = OFF_EMBR + Kc * 33 * Dc;   // lc_new      K x 31
constexpr long OFF_H    = OFF_LC + Kc * 31;          // h_new       K x 31
constexpr long OFF_EMBT = OFF_H + Kc * 31;           // embt_new    K x 31 x 64
constexpr long OFF_LF   = OFF_EMBT + Kc * 31 * Dc;   // lf_new      K x 31 x 256 x 4
constexpr long OFF_LPI  = OFF_LF + (long)Kc * 31 * Sc * Ac; // log_pi_new  K
constexpr long OFF_LW   = OFF_LPI + Kc;              // log_weight  K
constexpr long OFF_LL   = OFF_LW + Kc;               // log_lik     K

// Workspace offsets (floats)
constexpr long WS_LOGP1 = 0;                          // Z*16
constexpr long WS_LOGP2 = WS_LOGP1 + Zc * 16;         // Z*16
constexpr long WS_LSTAT = WS_LOGP2 + Zc * 16;         // (K*T+Z)*4
constexpr long WS_NEWLF = WS_LSTAT + (Kc * Tc + Zc) * 4; // Z*1024
constexpr long WS_LLM   = WS_NEWLF + (long)Zc * Sc * Ac; // Z
constexpr long WS_LLO   = WS_LLM + Zc;                // K*T
constexpr long WS_LOGW  = WS_LLO + Kc * Tc;           // Z
constexpr long WS_LPN   = WS_LOGW + Zc;               // Z
constexpr long WS_LLIK  = WS_LPN + Zc;                // Z
constexpr long WS_SELJ  = WS_LLIK + Zc;               // K (ints)

typedef __attribute__((ext_vector_type(16))) _Float16 v16h;
typedef __attribute__((ext_vector_type(8)))  float    v8f;

union H16 { v16h v; _Float16 e[16]; };

__device__ inline float softplusf(float x) {
    // numerically stable softplus, matches jax.nn.softplus
    return (x > 0.f) ? x + log1pf(expf(-x)) : log1pf(expf(x));
}

__device__ inline float lse4(float a, float b, float c, float d) {
    float m = fmaxf(fmaxf(a, b), fmaxf(c, d));
    return m + logf(expf(a - m) + expf(b - m) + expf(c - m) + expf(d - m));
}

// 4x4 matrix exponential (scaling & squaring + Taylor-12), then log(clip(.,1e-30))
__device__ void expm_log4(const float Q[4][4], float t, float* __restrict__ out) {
    float A[4][4];
    float nrm = 0.f;
    #pragma unroll
    for (int i = 0; i < 4; ++i) {
        float r = 0.f;
        #pragma unroll
        for (int j = 0; j < 4; ++j) { A[i][j] = Q[i][j] * t; r += fabsf(A[i][j]); }
        nrm = fmaxf(nrm, r);
    }
    int s = 0;
    if (nrm > 0.25f) {
        s = (int)ceilf(log2f(nrm * 4.0f));
        if (s < 0) s = 0;
        if (s > 30) s = 30;
    }
    float sc = exp2f(-(float)s);
    #pragma unroll
    for (int i = 0; i < 4; ++i)
        #pragma unroll
        for (int j = 0; j < 4; ++j) A[i][j] *= sc;

    float E[4][4], Tm[4][4], Tn[4][4];
    #pragma unroll
    for (int i = 0; i < 4; ++i)
        #pragma unroll
        for (int j = 0; j < 4; ++j) {
            E[i][j] = ((i == j) ? 1.f : 0.f) + A[i][j];
            Tm[i][j] = A[i][j];
        }
    for (int k = 2; k <= 12; ++k) {
        float inv = 1.0f / (float)k;
        #pragma unroll
        for (int i = 0; i < 4; ++i)
            #pragma unroll
            for (int j = 0; j < 4; ++j) {
                float acc = 0.f;
                #pragma unroll
                for (int l = 0; l < 4; ++l) acc += Tm[i][l] * A[l][j];
                Tn[i][j] = acc * inv;
            }
        #pragma unroll
        for (int i = 0; i < 4; ++i)
            #pragma unroll
            for (int j = 0; j < 4; ++j) { Tm[i][j] = Tn[i][j]; E[i][j] += Tn[i][j]; }
    }
    for (int q = 0; q < s; ++q) {
        #pragma unroll
        for (int i = 0; i < 4; ++i)
            #pragma unroll
            for (int j = 0; j < 4; ++j) {
                float acc = 0.f;
                #pragma unroll
                for (int l = 0; l < 4; ++l) acc += E[i][l] * E[l][j];
                Tn[i][j] = acc;
            }
        #pragma unroll
        for (int i = 0; i < 4; ++i)
            #pragma unroll
            for (int j = 0; j < 4; ++j) E[i][j] = Tn[i][j];
    }
    #pragma unroll
    for (int i = 0; i < 4; ++i)
        #pragma unroll
        for (int j = 0; j < 4; ++j)
            out[i * 4 + j] = logf(fmaxf(E[i][j], 1e-30f));
}

// K1: Q logits via WMMA (Z x 16 = emb_Z[Zx64] @ W_q[64x16]), then per-row expm
// One wave32 per 16 rows of Z.
__global__ __launch_bounds__(32)
void k1_q_expm(const float* __restrict__ embJ,  // [Z][64]
               const float* __restrict__ Wq,    // [64][16]
               const float* __restrict__ br1,   // [Z]
               const float* __restrict__ br2,   // [Z]
               float* __restrict__ logP1,       // [Z][16]
               float* __restrict__ logP2)       // [Z][16]
{
    __shared__ float lds[16 * 16];
    const int lane = threadIdx.x;
    const int z0 = blockIdx.x * 16;
    const int m = lane & 15;
    const int hi = lane >> 4;

    v8f c = {};
    #pragma unroll
    for (int ch = 0; ch < 2; ++ch) {
        H16 a, b;
        #pragma unroll
        for (int h = 0; h < 16; ++h) {
            int ka = (h < 8) ? (hi * 8 + h) : (16 + hi * 8 + (h - 8)); // A frag K index
            a.e[h] = (_Float16)embJ[(z0 + m) * Dc + ch * 32 + ka];
            int kb = (h < 8) ? (hi * 8 + h) : (16 + hi * 8 + (h - 8)); // B frag K index
            b.e[h] = (_Float16)Wq[(ch * 32 + kb) * 16 + m];
        }
        c = __builtin_amdgcn_wmma_f32_16x16x32_f16(false, a.v, false, b.v,
                                                   (short)0, c, false, false);
    }
    #pragma unroll
    for (int v = 0; v < 8; ++v) lds[(v + 8 * hi) * 16 + m] = c[v];
    __syncthreads();

    if (lane < 16) {
        const int z = z0 + lane;
        float Q[4][4];
        float rs[4] = {0.f, 0.f, 0.f, 0.f};
        #pragma unroll
        for (int i = 0; i < 4; ++i)
            #pragma unroll
            for (int j = 0; j < 4; ++j) {
                if (i == j) { Q[i][j] = 0.f; continue; }
                float sp = softplusf(lds[lane * 16 + i * 4 + j]);
                Q[i][j] = sp;
                rs[i] += sp;
            }
        #pragma unroll
        for (int i = 0; i < 4; ++i) Q[i][i] = -rs[i];
        expm_log4(Q, br1[z], logP1 + z * 16);
        expm_log4(Q, br2[z], logP2 + z * 16);
    }
}

// K2: stationary-dist logits for all K*T original nodes + Z merged nodes.
// rows = K*T + Z = 4608 -> 288 tiles of 16. WMMA with W_stat padded N=4->16.
__global__ __launch_bounds__(32)
void k2_stat(const float* __restrict__ embT,   // [K*T][64]
             const float* __restrict__ embJ,   // [Z][64]
             const float* __restrict__ Wstat,  // [64][4]
             float* __restrict__ lstat)        // [(K*T+Z)][4] log_softmax
{
    __shared__ float lds[16 * 16];
    const int lane = threadIdx.x;
    const int r0 = blockIdx.x * 16;
    const int m = lane & 15;
    const int hi = lane >> 4;
    const int row = r0 + m;
    const float* rp = (row < Kc * Tc) ? (embT + (long)row * Dc)
                                      : (embJ + (long)(row - Kc * Tc) * Dc);
    v8f c = {};
    #pragma unroll
    for (int ch = 0; ch < 2; ++ch) {
        H16 a, b;
        #pragma unroll
        for (int h = 0; h < 16; ++h) {
            int ka = (h < 8) ? (hi * 8 + h) : (16 + hi * 8 + (h - 8));
            a.e[h] = (_Float16)rp[ch * 32 + ka];
            int kb = (h < 8) ? (hi * 8 + h) : (16 + hi * 8 + (h - 8));
            b.e[h] = (m < Ac) ? (_Float16)Wstat[(ch * 32 + kb) * Ac + m] : (_Float16)0.f;
        }
        c = __builtin_amdgcn_wmma_f32_16x16x32_f16(false, a.v, false, b.v,
                                                   (short)0, c, false, false);
    }
    #pragma unroll
    for (int v = 0; v < 8; ++v) lds[(v + 8 * hi) * 16 + m] = c[v];
    __syncthreads();

    if (lane < 16) {
        const int r = r0 + lane;
        float l0 = lds[lane * 16 + 0], l1 = lds[lane * 16 + 1];
        float l2 = lds[lane * 16 + 2], l3 = lds[lane * 16 + 3];
        float lse = lse4(l0, l1, l2, l3);
        lstat[r * 4 + 0] = l0 - lse;
        lstat[r * 4 + 1] = l1 - lse;
        lstat[r * 4 + 2] = l2 - lse;
        lstat[r * 4 + 3] = l3 - lse;
    }
}

// K3: merged-node Felsenstein pruning + merged-node site likelihood sum.
// One block per candidate z, 256 threads = sites.
__global__ __launch_bounds__(256)
void k3_felsen(const int* __restrict__ indexes,
               const int* __restrict__ idx1, const int* __restrict__ idx2,
               const float* __restrict__ lf,      // [K][T][S][A]
               const float* __restrict__ logP1,   // [Z][16]
               const float* __restrict__ logP2,   // [Z][16]
               const float* __restrict__ lstat,   // [(K*T+Z)][4]
               float* __restrict__ newlf,         // [Z][S][A]
               float* __restrict__ llm)           // [Z]
{
    __shared__ float p1[16], p2[16], st[4];
    __shared__ float red[256];
    const int z = blockIdx.x;
    const int tid = threadIdx.x;
    if (tid < 16)       p1[tid] = logP1[z * 16 + tid];
    else if (tid < 32)  p2[tid - 16] = logP2[z * 16 + (tid - 16)];
    else if (tid < 36)  st[tid - 32] = lstat[(long)(Kc * Tc + z) * 4 + (tid - 32)];
    __syncthreads();

    const int k = z / Jc;
    const int kk = indexes[k];
    const int i1 = idx1[z], i2 = idx2[z];

    const float4 f1 = *(const float4*)(lf + ((long)(kk * Tc + i1) * Sc + tid) * Ac);
    const float4 f2 = *(const float4*)(lf + ((long)(kk * Tc + i2) * Sc + tid) * Ac);

    float nl[4];
    #pragma unroll
    for (int a = 0; a < 4; ++a) {
        float t1 = lse4(p1[a * 4 + 0] + f1.x, p1[a * 4 + 1] + f1.y,
                        p1[a * 4 + 2] + f1.z, p1[a * 4 + 3] + f1.w);
        float t2 = lse4(p2[a * 4 + 0] + f2.x, p2[a * 4 + 1] + f2.y,
                        p2[a * 4 + 2] + f2.z, p2[a * 4 + 3] + f2.w);
        nl[a] = t1 + t2;
    }
    float4 o; o.x = nl[0]; o.y = nl[1]; o.z = nl[2]; o.w = nl[3];
    *(float4*)(newlf + (long)z * Sc * Ac + tid * 4) = o;

    float site = lse4(nl[0] + st[0], nl[1] + st[1], nl[2] + st[2], nl[3] + st[3]);
    red[tid] = site;
    __syncthreads();
    for (int off = 128; off > 0; off >>= 1) {
        if (tid < off) red[tid] += red[tid + off];
        __syncthreads();
    }
    if (tid == 0) llm[z] = red[0];
}

// K4: per original (k,t) node likelihood: sum_s lse_a(lf + log_stat)
__global__ __launch_bounds__(256)
void k4_llorig(const float* __restrict__ lf,     // [K][T][S][A]
               const float* __restrict__ lstat,  // rows 0..K*T-1
               float* __restrict__ llo)          // [K*T]
{
    __shared__ float red[256];
    const int g = blockIdx.x;       // k*T + t
    const int tid = threadIdx.x;
    const float4 f = *(const float4*)(lf + ((long)g * Sc + tid) * Ac);
    float s0 = lstat[g * 4 + 0], s1 = lstat[g * 4 + 1];
    float s2 = lstat[g * 4 + 2], s3 = lstat[g * 4 + 3];
    red[tid] = lse4(f.x + s0, f.y + s1, f.z + s2, f.w + s3);
    __syncthreads();
    for (int off = 128; off > 0; off >>= 1) {
        if (tid < off) red[tid] += red[tid + off];
        __syncthreads();
    }
    if (tid == 0) llo[g] = red[0];
}

// K5: per-candidate log weights (priors, topo, v_minus), one thread per z
__global__ __launch_bounds__(256)
void k5_weights(const int* __restrict__ indexes,
                const int* __restrict__ idx1, const int* __restrict__ idx2,
                const int* __restrict__ leafc,   // [K][T]
                const float* __restrict__ b1len, const float* __restrict__ b2len, // [K][R]
                const float* __restrict__ br1, const float* __restrict__ br2,     // [Z]
                const float* __restrict__ logpi, // [K]
                const float* __restrict__ logvp, // [Z]
                const float* __restrict__ logdf, // [2N]
                const float* __restrict__ llo, const float* __restrict__ llm,
                float* __restrict__ logw, float* __restrict__ lpnew,
                float* __restrict__ llik)
{
    const int z = blockIdx.x * blockDim.x + threadIdx.x;
    if (z >= Zc) return;
    const int k = z / Jc;
    const int kk = indexes[k];
    const int i1 = idx1[z], i2 = idx2[z];

    // log-likelihood: precomputed node sums minus removed nodes plus merged node
    float sll = 0.f;
    for (int t = 0; t < Tc; ++t) sll += llo[kk * Tc + t];
    const float ll = sll - llo[kk * Tc + i1] - llo[kk * Tc + i2] + llm[z];

    // exponential branch prior over 33 + 33 branches
    float sb = 0.f;
    for (int r = 0; r < Rc; ++r) sb += b1len[kk * Rc + r] + b2len[kk * Rc + r];
    sb += br1[z] + br2[z];
    const float rate = 10.0f; // 1/PRIOR_BL
    const float lbp = 66.0f * logf(rate) - rate * sb;

    // topology prior + v_minus over the 31 nodes of the merged forest
    float lt = 0.f;
    int cnt = 0, lc1 = 0, lc2 = 0;
    for (int t = 0; t < Tc; ++t) {
        int cI = leafc[kk * Tc + t];
        if (t == i1) { lc1 = cI; continue; }
        if (t == i2) { lc2 = cI; continue; }
        int di = 2 * cI - 3;
        di = di < 0 ? 0 : (di > 2 * Nc - 1 ? 2 * Nc - 1 : di);
        lt -= logdf[di];
        if (cI > 1) cnt++;
    }
    {
        int cm = lc1 + lc2;
        int di = 2 * cm - 3;
        di = di < 0 ? 0 : (di > 2 * Nc - 1 ? 2 * Nc - 1 : di);
        lt -= logdf[di];
        cnt += 1;  // merged count >= 2 > 1 always
    }
    const float lpn = ll + lbp + lt;
    logw[z] = lpn - logpi[kk] + logf((float)cnt) - logvp[z];
    lpnew[z] = lpn;
    llik[z] = ll;
}

// K6: per-particle Gumbel selection + all small outputs
__global__ __launch_bounds__(128)
void k6_select(const int* __restrict__ indexes,
               const int* __restrict__ idx1, const int* __restrict__ idx2,
               const int* __restrict__ m1, const int* __restrict__ m2,    // [K][R]
               const float* __restrict__ b1len, const float* __restrict__ b2len, // [K][R]
               const float* __restrict__ br1, const float* __restrict__ br2,     // [Z]
               const float* __restrict__ embR,  // [K][R][D]
               const float* __restrict__ embT,  // [K][T][D]
               const float* __restrict__ embJ,  // [Z][D]
               const int* __restrict__ leafc, const int* __restrict__ hsh, // [K][T]
               const float* __restrict__ gumbel, // [Z]
               const float* __restrict__ logw, const float* __restrict__ lpnew,
               const float* __restrict__ llik,
               float* __restrict__ out, int* __restrict__ selj)
{
    const int k = blockIdx.x;
    __shared__ int sj;
    if (threadIdx.x == 0) {
        float w0 = logw[k * 4 + 0], w1 = logw[k * 4 + 1];
        float w2 = logw[k * 4 + 2], w3 = logw[k * 4 + 3];
        out[OFF_LW + k] = lse4(w0, w1, w2, w3) - logf((float)Jc);
        float s0 = w0 + gumbel[k * 4 + 0], s1 = w1 + gumbel[k * 4 + 1];
        float s2 = w2 + gumbel[k * 4 + 2], s3 = w3 + gumbel[k * 4 + 3];
        int best = 0; float bv = s0;
        if (s1 > bv) { bv = s1; best = 1; }
        if (s2 > bv) { bv = s2; best = 2; }
        if (s3 > bv) { bv = s3; best = 3; }
        sj = best;
        selj[k] = best;
    }
    __syncthreads();
    const int zs = k * Jc + sj;
    const int kk = indexes[k];
    const int i1 = idx1[zs], i2 = idx2[zs];
    const int a = i1 < i2 ? i1 : i2;
    const int b = i1 < i2 ? i2 : i1;

    for (int i = threadIdx.x; i < 33; i += blockDim.x) {
        out[OFF_M1 + (long)k * 33 + i] = (i < Rc) ? (float)m1[kk * Rc + i] : (float)i1;
        out[OFF_M2 + (long)k * 33 + i] = (i < Rc) ? (float)m2[kk * Rc + i] : (float)i2;
        out[OFF_B1 + (long)k * 33 + i] = (i < Rc) ? b1len[kk * Rc + i] : br1[zs];
        out[OFF_B2 + (long)k * 33 + i] = (i < Rc) ? b2len[kk * Rc + i] : br2[zs];
    }
    for (int u = threadIdx.x; u < 31; u += blockDim.x) {
        if (u < 30) {
            int s = u + (u >= a ? 1 : 0) + (u >= b - 1 ? 1 : 0);
            out[OFF_LC + (long)k * 31 + u] = (float)leafc[kk * Tc + s];
            out[OFF_H + (long)k * 31 + u] = (float)hsh[kk * Tc + s];
        } else {
            out[OFF_LC + (long)k * 31 + u] =
                (float)(leafc[kk * Tc + i1] + leafc[kk * Tc + i2]);
            int h1 = hsh[kk * Tc + i1], h2 = hsh[kk * Tc + i2];
            int mn = h1 < h2 ? h1 : h2;
            int mx = h1 < h2 ? h2 : h1;
            unsigned hv = ((unsigned)mn * 1000003u + (unsigned)mx) * 40503u + 2531011u;
            out[OFF_H + (long)k * 31 + u] = (float)(int)hv;
        }
    }
    for (int i = threadIdx.x; i < 33 * Dc; i += blockDim.x) {
        int r = i >> 6, d = i & 63;
        out[OFF_EMBR + (long)k * 33 * Dc + i] =
            (r < Rc) ? embR[(long)(kk * Rc + r) * Dc + d] : embJ[(long)zs * Dc + d];
    }
    for (int i = threadIdx.x; i < 31 * Dc; i += blockDim.x) {
        int u = i >> 6, d = i & 63;
        float v;
        if (u < 30) {
            int s = u + (u >= a ? 1 : 0) + (u >= b - 1 ? 1 : 0);
            v = embT[(long)(kk * Tc + s) * Dc + d];
        } else {
            v = embJ[(long)zs * Dc + d];
        }
        out[OFF_EMBT + (long)k * 31 * Dc + i] = v;
    }
    if (threadIdx.x == 0) {
        out[OFF_LPI + k] = lpnew[zs];
        out[OFF_LL + k] = llik[zs];
    }
}

// K7: lf_new output for the selected candidate: (K, 31) blocks x 256 threads (float4 each)
__global__ __launch_bounds__(256)
void k7_lfout(const int* __restrict__ indexes,
              const int* __restrict__ idx1, const int* __restrict__ idx2,
              const int* __restrict__ selj,
              const float* __restrict__ lf,     // [K][T][S][A]
              const float* __restrict__ newlf,  // [Z][S][A]
              float* __restrict__ out)
{
    const int k = blockIdx.x, u = blockIdx.y, tid = threadIdx.x;
    const int zs = k * Jc + selj[k];
    const int kk = indexes[k];
    const int i1 = idx1[zs], i2 = idx2[zs];
    const int a = i1 < i2 ? i1 : i2;
    const int b = i1 < i2 ? i2 : i1;
    const float* srcp;
    if (u < 30) {
        int s = u + (u >= a ? 1 : 0) + (u >= b - 1 ? 1 : 0);
        srcp = lf + (long)(kk * Tc + s) * Sc * Ac;
    } else {
        srcp = newlf + (long)zs * Sc * Ac;
    }
    float4 v = *(const float4*)(srcp + tid * 4);
    *(float4*)(out + OFF_LF + ((long)(k * 31 + u)) * Sc * Ac + tid * 4) = v;
}

extern "C" void kernel_launch(void* const* d_in, const int* in_sizes, int n_in,
                              void* d_out, int out_size, void* d_ws, size_t ws_size,
                              hipStream_t stream) {
    (void)in_sizes; (void)n_in; (void)out_size; (void)ws_size;
    const int*   indexes = (const int*)d_in[0];
    const int*   m1      = (const int*)d_in[1];
    const int*   m2      = (const int*)d_in[2];
    const float* b1len   = (const float*)d_in[3];
    const float* b2len   = (const float*)d_in[4];
    const float* embR    = (const float*)d_in[5];
    const int*   leafc   = (const int*)d_in[6];
    const int*   hsh     = (const int*)d_in[7];
    const float* embT    = (const float*)d_in[8];
    const float* lf      = (const float*)d_in[9];
    const float* logpi   = (const float*)d_in[10];
    const int*   idx1    = (const int*)d_in[11];
    const int*   idx2    = (const int*)d_in[12];
    const float* br1     = (const float*)d_in[13];
    const float* br2     = (const float*)d_in[14];
    const float* embJ    = (const float*)d_in[15];
    const float* logvp   = (const float*)d_in[16];
    const float* gumbel  = (const float*)d_in[17];
    const float* Wq      = (const float*)d_in[18];
    const float* Wstat   = (const float*)d_in[19];
    const float* logdf   = (const float*)d_in[20];

    float* w     = (float*)d_ws;
    float* logP1 = w + WS_LOGP1;
    float* logP2 = w + WS_LOGP2;
    float* lstat = w + WS_LSTAT;
    float* newlf = w + WS_NEWLF;
    float* llm   = w + WS_LLM;
    float* llo   = w + WS_LLO;
    float* logw  = w + WS_LOGW;
    float* lpnew = w + WS_LPN;
    float* llik  = w + WS_LLIK;
    int*   selj  = (int*)(w + WS_SELJ);

    k1_q_expm<<<Zc / 16, 32, 0, stream>>>(embJ, Wq, br1, br2, logP1, logP2);
    k2_stat<<<(Kc * Tc + Zc) / 16, 32, 0, stream>>>(embT, embJ, Wstat, lstat);
    k3_felsen<<<Zc, 256, 0, stream>>>(indexes, idx1, idx2, lf, logP1, logP2,
                                      lstat, newlf, llm);
    k4_llorig<<<Kc * Tc, 256, 0, stream>>>(lf, lstat, llo);
    k5_weights<<<(Zc + 255) / 256, 256, 0, stream>>>(indexes, idx1, idx2, leafc,
                                                     b1len, b2len, br1, br2, logpi,
                                                     logvp, logdf, llo, llm,
                                                     logw, lpnew, llik);
    k6_select<<<Kc, 128, 0, stream>>>(indexes, idx1, idx2, m1, m2, b1len, b2len,
                                      br1, br2, embR, embT, embJ, leafc, hsh,
                                      gumbel, logw, lpnew, llik, (float*)d_out, selj);
    k7_lfout<<<dim3(Kc, 31), 256, 0, stream>>>(indexes, idx1, idx2, selj, lf,
                                               newlf, (float*)d_out);
}
